// ImageAdaptive3DLUT_6322191860245
// MI455X (gfx1250) — matrix operand order repaired and checked by reading, not compiled, not run on hardware
//
#include <hip/hip_runtime.h>
#include <hip/hip_bf16.h>
#include <math.h>

typedef __attribute__((ext_vector_type(16))) _Float16 v16h;
typedef __attribute__((ext_vector_type(8)))  float    v8f;

typedef __attribute__((ext_vector_type(4))) unsigned int tdm_u32x4;
typedef __attribute__((ext_vector_type(8))) int          tdm_i32x8;
typedef __attribute__((ext_vector_type(4))) int          tdm_i32x4;

#define LUTS  33
#define LUT3  35937           // 33^3
#define IMG_H 1080
#define IMG_W 1920
#define HW    (IMG_H * IMG_W)
#define NB    4

// ---------------------------------------------------------------------------
// 1) Anti-aliased bilinear resize [4,3,1080,1920] -> [4,3,256,256]
//    Matches jax.image.resize(method='linear', antialias=True).
// ---------------------------------------------------------------------------
__global__ void k_resize(const float* __restrict__ img, float* __restrict__ thumb) {
    int i = blockIdx.x * blockDim.x + threadIdx.x;
    if (i >= NB * 3 * 256 * 256) return;
    int ox = i & 255;
    int oy = (i >> 8) & 255;
    int bc = i >> 16;                     // 0..11 = b*3+c plane
    const float sy = (float)IMG_H / 256.0f;   // 4.21875
    const float sx = (float)IMG_W / 256.0f;   // 7.5
    float cy = (oy + 0.5f) * sy - 0.5f;
    float cx = (ox + 0.5f) * sx - 0.5f;
    int y0 = (int)ceilf(cy - sy);  if (y0 < 0) y0 = 0;
    int y1 = (int)floorf(cy + sy); if (y1 > IMG_H - 1) y1 = IMG_H - 1;
    int x0 = (int)ceilf(cx - sx);  if (x0 < 0) x0 = 0;
    int x1 = (int)floorf(cx + sx); if (x1 > IMG_W - 1) x1 = IMG_W - 1;
    const float* src = img + (size_t)bc * HW;

    float wxs = 0.f;
    for (int x = x0; x <= x1; ++x)
        wxs += fmaxf(1.f - fabsf((float)x - cx) / sx, 0.f);

    float wys = 0.f, num = 0.f;
    for (int y = y0; y <= y1; ++y) {
        float wy = fmaxf(1.f - fabsf((float)y - cy) / sy, 0.f);
        if (wy <= 0.f) continue;
        wys += wy;
        const float* row = src + (size_t)y * IMG_W;
        float acc = 0.f;
        for (int x = x0; x <= x1; ++x) {
            float wx = fmaxf(1.f - fabsf((float)x - cx) / sx, 0.f);
            acc += wx * row[x];
        }
        num += wy * acc;
    }
    thumb[i] = num / (wys * wxs);
}

// ---------------------------------------------------------------------------
// 2) conv1 as WMMA GEMM: [4,3,256,256] -> relu -> [4,16,128,128]
//    M=16 out-channels, N=16 pixels per wave, K=27 padded to 32.
// ---------------------------------------------------------------------------
__global__ void k_conv1(const float* __restrict__ in,    // thumb
                        const float* __restrict__ w,     // [16,3,3,3]
                        const float* __restrict__ bias,  // [16]
                        float* __restrict__ out) {       // [4,16,128,128]
    int tile = blockIdx.x;        // NB*1024 tiles
    int b    = tile >> 10;
    int tp   = tile & 1023;
    int p0   = tp * 16;
    int lane = threadIdx.x;       // 0..31
    int m    = lane & 15;
    int kg   = lane >> 4;

    v16h a;
#pragma unroll
    for (int e = 0; e < 16; ++e) {
        int k = e + 16 * kg;
        a[e] = (k < 27) ? (_Float16)w[m * 27 + k] : (_Float16)0.f;
    }

    int n  = lane & 15;
    int p  = p0 + n;
    int oy = p >> 7, ox = p & 127;
    v16h bm;
#pragma unroll
    for (int e = 0; e < 16; ++e) {
        int k = e + 16 * kg;
        float v = 0.f;
        if (k < 27) {
            int c = k / 9, r = k % 9, ky = r / 3, kx = r % 3;
            int iy = 2 * oy + ky, ix = 2 * ox + kx;
            if (iy < 256 && ix < 256)
                v = in[((size_t)(b * 3 + c) * 256 + iy) * 256 + ix];
        }
        bm[e] = (_Float16)v;
    }

    v8f acc = {};
    acc = __builtin_amdgcn_wmma_f32_16x16x32_f16(false, a, false, bm,
                                                 (short)0, acc, false, false);
#pragma unroll
    for (int r = 0; r < 8; ++r) {
        int M = r + 8 * kg;
        float val = acc[r] + bias[M];
        val = val > 0.f ? val : 0.f;
        out[(size_t)(b * 16 + M) * (128 * 128) + p] = val;
    }
}

// ---------------------------------------------------------------------------
// 3) conv2 as WMMA GEMM: [4,16,128,128] -> relu -> [4,32,64,64]
//    M=32 (two 16-row M-tiles), K=144 padded to 160 (5 WMMA steps).
// ---------------------------------------------------------------------------
__global__ void k_conv2(const float* __restrict__ in,    // conv1 out
                        const float* __restrict__ w,     // [32,16,3,3]
                        const float* __restrict__ bias,  // [32]
                        float* __restrict__ out) {       // [4,32,64,64]
    int id   = blockIdx.x;        // NB * 2 * 256
    int b    = id >> 9;
    int rem  = id & 511;
    int mt   = rem >> 8;
    int tp   = rem & 255;
    int p0   = tp * 16;
    int lane = threadIdx.x;
    int kg   = lane >> 4;
    int mrow = (lane & 15) + mt * 16;
    int n    = lane & 15;
    int p    = p0 + n;
    int oy   = p >> 6, ox = p & 63;

    v8f acc = {};
#pragma unroll
    for (int kc = 0; kc < 5; ++kc) {
        int kbase = kc * 32;
        v16h a, bm;
#pragma unroll
        for (int e = 0; e < 16; ++e) {
            int k = kbase + e + 16 * kg;
            a[e] = (k < 144) ? (_Float16)w[mrow * 144 + k] : (_Float16)0.f;
        }
#pragma unroll
        for (int e = 0; e < 16; ++e) {
            int k = kbase + e + 16 * kg;
            float v = 0.f;
            if (k < 144) {
                int c = k / 9, r = k % 9, ky = r / 3, kx = r % 3;
                int iy = 2 * oy + ky, ix = 2 * ox + kx;
                if (iy < 128 && ix < 128)
                    v = in[((size_t)(b * 16 + c) * 128 + iy) * 128 + ix];
            }
            bm[e] = (_Float16)v;
        }
        acc = __builtin_amdgcn_wmma_f32_16x16x32_f16(false, a, false, bm,
                                                     (short)0, acc, false, false);
    }
#pragma unroll
    for (int r = 0; r < 8; ++r) {
        int M = r + 8 * kg + mt * 16;
        float val = acc[r] + bias[M];
        val = val > 0.f ? val : 0.f;
        out[(size_t)(b * 32 + M) * 4096 + p] = val;
    }
}

// ---------------------------------------------------------------------------
// 4) Global average pool via Tensor Data Mover: each wave DMAs its 16 KB
//    plane (4096 f32) into LDS with tensor_load_to_lds, waits on TENSORcnt,
//    then reduces with a wave32 shuffle tree. One wave per (b,c).
//    D# packing per CDNA5 ISA:
//      g0: count=1 | lds_addr | global_addr[56:0] | type=2
//      g1: data_size=2 (4B); tensor_dim0=4096; tensor_dim1=1;
//          tile_dim0=4096; tile_dim1=1; tensor_dim0_stride=4096
// ---------------------------------------------------------------------------
__global__ void k_pool(const float* __restrict__ in, float* __restrict__ pooled) {
    __shared__ float buf[4096];
    int bc = blockIdx.x;                  // b*32 + c, 128 blocks
    const float* src = in + (size_t)bc * 4096;

    unsigned long long ga = (unsigned long long)(uintptr_t)src;
    unsigned int lds_addr = (unsigned int)(uintptr_t)(&buf[0]);

    tdm_u32x4 g0 = {
        1u,                                        // count=1 (valid, user mode)
        lds_addr,                                  // lds_addr (bytes)
        (unsigned int)(ga & 0xFFFFFFFFu),          // global_addr[31:0]
        (unsigned int)((ga >> 32) & 0x01FFFFFFu)   // global_addr[56:32]
            | (2u << 30)                           // type=2 ("image")
    };
    tdm_i32x8 g1 = {
        (int)(2u << 16),        // d0: workgroup_mask=0, data_size=2 (4B)
        (int)(4096u << 16),     // d1: atomic_barrier=0, tensor_dim0[15:0]=4096
        (int)(1u << 16),        // d2: tensor_dim0[31:16]=0, tensor_dim1[15:0]=1
        (int)(4096u << 16),     // d3: tensor_dim1[31:16]=0, tile_dim0=4096
        (int)(1u),              // d4: tile_dim1=1, tile_dim2=0
        (int)(4096u),           // d5: tensor_dim0_stride[31:0]=4096
        0,                      // d6: stride hi / tensor_dim1_stride lo
        0                       // d7: tensor_dim1_stride hi
    };
    tdm_i32x4 gz = {0, 0, 0, 0};

#if __clang_major__ >= 23
    tdm_i32x8 gz8 = {0, 0, 0, 0, 0, 0, 0, 0};
    __builtin_amdgcn_tensor_load_to_lds(g0, g1, gz, gz, gz8, 0);
#else
    __builtin_amdgcn_tensor_load_to_lds(g0, g1, gz, gz, 0);
#endif
    __builtin_amdgcn_s_wait_tensorcnt(0);

    float s = 0.f;
    for (int i = threadIdx.x; i < 4096; i += 32) s += buf[i];
#pragma unroll
    for (int off = 16; off > 0; off >>= 1) s += __shfl_down(s, off, 32);
    if (threadIdx.x == 0) pooled[bc] = s * (1.0f / 4096.0f);
}

// ---------------------------------------------------------------------------
// 5) Dense [32,3] + softmax -> per-batch basis weights [4,3]
// ---------------------------------------------------------------------------
__global__ void k_head(const float* __restrict__ pooled,
                       const float* __restrict__ dw,   // [32,3]
                       const float* __restrict__ db,   // [3]
                       float* __restrict__ wts) {      // [4,3]
    int b = threadIdx.x;
    if (b >= NB) return;
    float l[3];
#pragma unroll
    for (int n = 0; n < 3; ++n) {
        float s = db[n];
        for (int c = 0; c < 32; ++c) s += pooled[b * 32 + c] * dw[c * 3 + n];
        l[n] = s;
    }
    float m = fmaxf(l[0], fmaxf(l[1], l[2]));
    float e0 = expf(l[0] - m), e1 = expf(l[1] - m), e2 = expf(l[2] - m);
    float inv = 1.0f / (e0 + e1 + e2);
    wts[b * 3 + 0] = e0 * inv;
    wts[b * 3 + 1] = e1 * inv;
    wts[b * 3 + 2] = e2 * inv;
}

// ---------------------------------------------------------------------------
// 6) Blend basis LUTs: lut[b] = sum_n wts[b,n] * basis[n]   ([4,35937,3])
// ---------------------------------------------------------------------------
__global__ void k_blend(const float* __restrict__ basis,
                        const float* __restrict__ wts,
                        float* __restrict__ lut) {
    const int per_b = LUT3 * 3;
    int i = blockIdx.x * blockDim.x + threadIdx.x;
    if (i >= NB * per_b) return;
    int b  = i / per_b;
    int pc = i - b * per_b;
    lut[i] = wts[b * 3 + 0] * basis[pc]
           + wts[b * 3 + 1] * basis[per_b + pc]
           + wts[b * 3 + 2] * basis[2 * per_b + pc];
}

// ---------------------------------------------------------------------------
// 7) Trilinear 3D-LUT application, 4 pixels/thread so streaming traffic
//    moves as b128 (dominant kernel; ~200 MB vs 23.3 TB/s HBM; LUT gathers
//    hit L2 since 1.7 MB total << 192 MB).
// ---------------------------------------------------------------------------
__device__ __forceinline__ void lut_pixel(const float* __restrict__ L,
                                          float R, float G, float B,
                                          float& r, float& g, float& bb) {
    R = fminf(fmaxf(R, 0.f), 1.f) * (float)(LUTS - 1);
    G = fminf(fmaxf(G, 0.f), 1.f) * (float)(LUTS - 1);
    B = fminf(fmaxf(B, 0.f), 1.f) * (float)(LUTS - 1);
    int x0 = (int)floorf(R); x0 = x0 < 0 ? 0 : (x0 > LUTS - 2 ? LUTS - 2 : x0);
    int y0 = (int)floorf(G); y0 = y0 < 0 ? 0 : (y0 > LUTS - 2 ? LUTS - 2 : y0);
    int z0 = (int)floorf(B); z0 = z0 < 0 ? 0 : (z0 > LUTS - 2 ? LUTS - 2 : z0);
    float fx = R - (float)x0, fy = G - (float)y0, fz = B - (float)z0;
    int x1 = x0 + 1, y1 = y0 + 1, z1 = z0 + 1;
    r = 0.f; g = 0.f; bb = 0.f;
#define CORNER(Z, Y, X, WT)                                         \
    {                                                               \
        int idx = (((Z) * LUTS + (Y)) * LUTS + (X)) * 3;            \
        float wt = (WT);                                            \
        r  += wt * L[idx];                                          \
        g  += wt * L[idx + 1];                                      \
        bb += wt * L[idx + 2];                                      \
    }
    float gx0 = 1.f - fx, gy0 = 1.f - fy, gz0 = 1.f - fz;
    CORNER(z0, y0, x0, gz0 * gy0 * gx0)
    CORNER(z0, y0, x1, gz0 * gy0 * fx)
    CORNER(z0, y1, x0, gz0 * fy  * gx0)
    CORNER(z0, y1, x1, gz0 * fy  * fx)
    CORNER(z1, y0, x0, fz  * gy0 * gx0)
    CORNER(z1, y0, x1, fz  * gy0 * fx)
    CORNER(z1, y1, x0, fz  * fy  * gx0)
    CORNER(z1, y1, x1, fz  * fy  * fx)
#undef CORNER
    r  = fminf(fmaxf(r, 0.f), 1.f);
    g  = fminf(fmaxf(g, 0.f), 1.f);
    bb = fminf(fmaxf(bb, 0.f), 1.f);
}

__global__ void k_apply(const float* __restrict__ img,
                        const float* __restrict__ lut,
                        float* __restrict__ out) {
    const int QP = HW / 4;                // quads per plane
    int i = blockIdx.x * blockDim.x + threadIdx.x;
    if (i >= NB * QP) return;
    int b = i / QP;
    int p = (i - b * QP) * 4;
    const float* ib = img + (size_t)b * 3 * HW;

    float4 R4 = *(const float4*)(ib + p);
    float4 G4 = *(const float4*)(ib + HW + p);
    float4 B4 = *(const float4*)(ib + 2 * HW + p);
    const float* L = lut + (size_t)b * LUT3 * 3;

    // gfx1250 prefetch of the gather neighborhood (global_prefetch_b8)
    {
        int zi = (int)(fminf(fmaxf(B4.x, 0.f), 1.f) * (float)(LUTS - 1));
        __builtin_prefetch(&L[zi * LUTS * LUTS * 3], 0, 3);
    }

    float rr[4], gg[4], bb[4];
    float Rs[4] = {R4.x, R4.y, R4.z, R4.w};
    float Gs[4] = {G4.x, G4.y, G4.z, G4.w};
    float Bs[4] = {B4.x, B4.y, B4.z, B4.w};
#pragma unroll
    for (int j = 0; j < 4; ++j)
        lut_pixel(L, Rs[j], Gs[j], Bs[j], rr[j], gg[j], bb[j]);

    float* ob = out + (size_t)b * 3 * HW;
    *(float4*)(ob + p)          = make_float4(rr[0], rr[1], rr[2], rr[3]);
    *(float4*)(ob + HW + p)     = make_float4(gg[0], gg[1], gg[2], gg[3]);
    *(float4*)(ob + 2 * HW + p) = make_float4(bb[0], bb[1], bb[2], bb[3]);
}

// ---------------------------------------------------------------------------
extern "C" void kernel_launch(void* const* d_in, const int* in_sizes, int n_in,
                              void* d_out, int out_size, void* d_ws, size_t ws_size,
                              hipStream_t stream) {
    (void)in_sizes; (void)n_in; (void)out_size; (void)ws_size;
    const float* image = (const float*)d_in[0];   // [4,3,1080,1920]
    const float* basis = (const float*)d_in[1];   // [3,35937,3]
    const float* c1w   = (const float*)d_in[2];   // [16,3,3,3]
    const float* c1b   = (const float*)d_in[3];   // [16]
    const float* c2w   = (const float*)d_in[4];   // [32,16,3,3]
    const float* c2b   = (const float*)d_in[5];   // [32]
    const float* dw    = (const float*)d_in[6];   // [32,3]
    const float* db    = (const float*)d_in[7];   // [3]
    float* out = (float*)d_out;

    float* ws     = (float*)d_ws;
    float* thumb  = ws;                                   //  786432 f
    float* c1o    = thumb  + (size_t)NB * 3 * 256 * 256;  // 1048576 f
    float* c2o    = c1o    + (size_t)NB * 16 * 128 * 128; //  524288 f
    float* pooled = c2o    + (size_t)NB * 32 * 64 * 64;   //     128 f
    float* wts    = pooled + 128;                         //      12 f
    float* lut    = wts    + 16;                          //  431244 f

    k_resize<<<(NB * 3 * 256 * 256 + 255) / 256, 256, 0, stream>>>(image, thumb);
    k_conv1 <<<NB * 1024, 32, 0, stream>>>(thumb, c1w, c1b, c1o);
    k_conv2 <<<NB * 512,  32, 0, stream>>>(c1o, c2w, c2b, c2o);
    k_pool  <<<NB * 32,   32, 0, stream>>>(c2o, pooled);
    k_head  <<<1, 32, 0, stream>>>(pooled, dw, db, wts);
    k_blend <<<(NB * LUT3 * 3 + 255) / 256, 256, 0, stream>>>(basis, wts, lut);
    k_apply <<<(NB * HW / 4 + 255) / 256, 256, 0, stream>>>(image, lut, out);
}
